// Denoising_69303592288398
// MI455X (gfx1250) — matrix operand order repaired and checked by reading, not compiled, not run on hardware
//
#include <hip/hip_runtime.h>
#include <hip/hip_bf16.h>
#include <math.h>

// Problem constants (from reference): B=2, C=64, T=8, H=W=48
#define Bn   2
#define Tn   8
#define Cn   64
#define C2n  32
#define Pn   2304        // 48*48
#define SLn  16          // B*T slices
#define NIT  (Pn / 32)   // 72 q-chunks

typedef __attribute__((ext_vector_type(16))) _Float16 v16h;
typedef __attribute__((ext_vector_type(8)))  float    v8f;

// ---------------------------------------------------------------------------
// Operand gathers in contiguous k order (element s: k = 16*(l>>4) + s), shared by A and B
// A-matrix f16 16x32 (M=16, K=32):
//   lane l: m = l&15 ; k = (l>>4)*16 + 2*v + h
//   VGPR v (0..7): k = (l>>4)*16 + 2*v + h
__device__ __forceinline__ v16h load_A(const _Float16* src, int rs, int lane) {
  const int m  = lane & 15;
  const int kh = (lane >> 4) * 16;
  union { v16h v; unsigned u[8]; } r;
  const unsigned* p = (const unsigned*)(src + m * rs);
#pragma unroll
  for (int vv = 0; vv < 8; ++vv) {
    const int k = kh + 2 * vv;
    r.u[vv] = p[k >> 1];
  }
  return r.v;
}

// B-matrix f16 32x16 (K=32, N=16), source laid out as [n][k] (k contiguous):
//   lane l: n = l&15 ; k = (l>>4)*16 + 2*v + h
__device__ __forceinline__ v16h load_B_ncontig(const _Float16* src, int rs,
                                               int kbase, int lane) {
  const int n  = lane & 15;
  const int kh = (lane >> 4) * 16;
  union { v16h v; unsigned u[8]; } r;
  const unsigned* p = (const unsigned*)(src + n * rs + kbase);
#pragma unroll
  for (int vv = 0; vv < 8; ++vv) r.u[vv] = p[(kh + 2 * vv) >> 1];
  return r.v;
}

// 16-lane reductions via DPP16 (pure VALU; no LDS traffic).
#define DPPF(x, ctrl)                                                     \
  __uint_as_float(__builtin_amdgcn_update_dpp(                            \
      0, (int)__float_as_uint(x), (ctrl), 0xf, 0xf, true))

__device__ __forceinline__ float rowmax16(float v) {
  v = fmaxf(v, DPPF(v, 0xB1));   // quad_perm [1,0,3,2]
  v = fmaxf(v, DPPF(v, 0x4E));   // quad_perm [2,3,0,1]
  v = fmaxf(v, DPPF(v, 0x141));  // row_half_mirror
  v = fmaxf(v, DPPF(v, 0x140));  // row_mirror
  return v;
}
__device__ __forceinline__ float rowsum16(float v) {
  v += DPPF(v, 0xB1);
  v += DPPF(v, 0x4E);
  v += DPPF(v, 0x141);
  v += DPPF(v, 0x140);
  return v;
}

// ---------------------------------------------------------------------------
// Kernel 1: G[t] = theta_w[t]^T * phi_w[t] / sqrt(C2)   (8 x 64x64 f32)
__global__ __launch_bounds__(256) void g_kernel(const float* __restrict__ tw,
                                                const float* __restrict__ pw,
                                                float* __restrict__ g) {
  const int t = blockIdx.x;
  const float scale = 0.17677669529663687f;  // 1/sqrt(32)
  for (int e = threadIdx.x; e < Cn * Cn; e += 256) {
    const int i = e >> 6, j = e & 63;
    float acc = 0.f;
#pragma unroll
    for (int a = 0; a < C2n; ++a)
      acc += tw[(t * C2n + a) * Cn + i] * pw[(t * C2n + a) * Cn + j];
    g[t * Cn * Cn + e] = acc * scale;
  }
}

// Kernel 2: stage xh[s][p][c] (f16) and yh = G x (f16), p-major / c-contig
__global__ __launch_bounds__(256) void xy_kernel(const float* __restrict__ x_in,
                                                 const float* __restrict__ g,
                                                 _Float16* __restrict__ xh,
                                                 _Float16* __restrict__ yh) {
  const int s = blockIdx.y;                       // slice = b*T + t
  const int p = blockIdx.x * 256 + threadIdx.x;   // 0..2303
  const int b = s >> 3, t = s & 7;
  float xc[Cn];
  const float* xb = x_in + ((size_t)(b * Cn) * Tn + t) * Pn + p;
#pragma unroll
  for (int c = 0; c < Cn; ++c) xc[c] = xb[(size_t)c * Tn * Pn];
  _Float16* xo = xh + ((size_t)s * Pn + p) * Cn;
#pragma unroll
  for (int c = 0; c < Cn; ++c) xo[c] = (_Float16)xc[c];
  const float* gt = g + t * Cn * Cn;
  _Float16* yo = yh + ((size_t)s * Pn + p) * Cn;
  for (int c = 0; c < Cn; ++c) {
    float acc = 0.f;
#pragma unroll
    for (int cc = 0; cc < Cn; ++cc) acc += gt[c * Cn + cc] * xc[cc];
    yo[c] = (_Float16)acc;
  }
}

// ---------------------------------------------------------------------------
// Kernel 3: fused flash attention + output projection + residual.
// Block = 256 threads = 8 waves; wave w owns rows p0..p0+15 of P.
// Double-buffered tiles: async copy of tile i+1 overlaps compute of tile i.
__global__ __launch_bounds__(256) void flash_kernel(
    const _Float16* __restrict__ xh, const _Float16* __restrict__ yh,
    const float* __restrict__ x_in, const float* __restrict__ out_w,
    float* __restrict__ out) {
  __shared__ _Float16 ows[Cn * Cn];          // out_w f16 [o][c]
  __shared__ _Float16 ytile[2][32 * Cn];     // y chunk [q][c] (async-filled)
  __shared__ _Float16 xtile_t[2][Cn * 32];   // x chunk [c][k'] (perm-transposed)
  __shared__ _Float16 stage[8][16 * 64];     // per-wave layout-shuffle buffer

  const int tid  = threadIdx.x;
  const int wave = tid >> 5;
  const int lane = tid & 31;
  const int s    = blockIdx.y;
  const int b    = s >> 3, t = s & 7;
  const int p0   = blockIdx.x * 128 + wave * 16;

  for (int e = tid; e < Cn * Cn; e += 256) ows[e] = (_Float16)out_w[e];

  // A operands for S = x^T y : A[m,k] = xh[p0+m][k], K = 64 in two chunks
  const _Float16* xrow = xh + ((size_t)s * Pn + p0) * Cn;
  const v16h Ax0 = load_A(xrow, Cn, lane);
  const v16h Ax1 = load_A(xrow + 32, Cn, lane);

  v8f acc[4];
#pragma unroll
  for (int j = 0; j < 4; ++j) acc[j] = (v8f){0.f,0.f,0.f,0.f,0.f,0.f,0.f,0.f};
  float mrow[8], lrow[8];
#pragma unroll
  for (int r = 0; r < 8; ++r) { mrow[r] = -1.0e30f; lrow[r] = 0.f; }

  const int q  = tid >> 3;              // cooperative-load coords
  const int c0 = (tid & 7) * 8;
  const int qp = 2 * (q & 15) + (q >> 4);  // permuted K index (subtile-interleave)

  // Issue tile `it` into buffer `buf`: y via CDNA5 async global->LDS copy,
  // x via register path (needs transpose + K-permutation on the way in).
  auto issue_tile = [&](int it, int buf) {
    const size_t rowi = ((size_t)s * Pn + it * 32 + q) * Cn + c0;
    const _Float16* ygp = yh + rowi;
    const unsigned ydst = (unsigned)(uintptr_t)(&ytile[buf][q * Cn + c0]);
    asm volatile("global_load_async_to_lds_b128 %0, %1, off"
                 :: "v"(ydst), "v"(ygp) : "memory");
    union { float4 f; _Float16 h[8]; } xv;
    xv.f = *(const float4*)(xh + rowi);
#pragma unroll
    for (int i = 0; i < 8; ++i) xtile_t[buf][(c0 + i) * 32 + qp] = xv.h[i];
    if (it + 1 < NIT)
      __builtin_prefetch(xh + rowi + (size_t)32 * Cn, 0, 1);
  };

  issue_tile(0, 0);  // prologue

  for (int it = 0; it < NIT; ++it) {
    const int cur = it & 1;
    __syncthreads();  // buffer (it+1)&1 fully consumed (previous iteration)
    if (it + 1 < NIT) {
      issue_tile(it + 1, cur ^ 1);
      asm volatile("s_wait_asynccnt 1" ::: "memory");  // tile `it` landed
    } else {
      asm volatile("s_wait_asynccnt 0" ::: "memory");
    }
    __syncthreads();  // tile `it` visible to all waves

    const _Float16* yt = ytile[cur];
    const _Float16* xt = xtile_t[cur];

    // S tiles: two 16x16 q-subtiles, K=64 (2 WMMAs each)
    v8f s0 = (v8f){0.f,0.f,0.f,0.f,0.f,0.f,0.f,0.f};
    v8f s1 = (v8f){0.f,0.f,0.f,0.f,0.f,0.f,0.f,0.f};
    s0 = __builtin_amdgcn_wmma_f32_16x16x32_f16(false, Ax0, false,
        load_B_ncontig(yt, Cn, 0, lane), (short)0, s0, false, false);
    s0 = __builtin_amdgcn_wmma_f32_16x16x32_f16(false, Ax1, false,
        load_B_ncontig(yt, Cn, 32, lane), (short)0, s0, false, false);
    s1 = __builtin_amdgcn_wmma_f32_16x16x32_f16(false, Ax0, false,
        load_B_ncontig(yt + 16 * Cn, Cn, 0, lane), (short)0, s1, false, false);
    s1 = __builtin_amdgcn_wmma_f32_16x16x32_f16(false, Ax1, false,
        load_B_ncontig(yt + 16 * Cn, Cn, 32, lane), (short)0, s1, false, false);

    // online softmax over this 32-column block (DPP16 reductions)
#pragma unroll
    for (int r = 0; r < 8; ++r) {
      const float mx = rowmax16(fmaxf(s0[r], s1[r]));
      const float mn = fmaxf(mrow[r], mx);
      const float alpha = __expf(mrow[r] - mn);
      const float e0 = __expf(s0[r] - mn);
      const float e1 = __expf(s1[r] - mn);
      mrow[r] = mn;
      lrow[r] = lrow[r] * alpha + rowsum16(e0 + e1);
#pragma unroll
      for (int j = 0; j < 4; ++j) acc[j][r] *= alpha;
      // P~ staged with permuted K: subtile0 col n -> k'=2n, subtile1 -> 2n+1;
      // the pair is adjacent -> single packed b32 store.
      const int row = r + ((lane >> 4) << 3);
      const int n = lane & 15;
      union { unsigned u; _Float16 h[2]; } pk;
      pk.h[0] = (_Float16)e0;
      pk.h[1] = (_Float16)e1;
      *(unsigned*)&stage[wave][row * 64 + 2 * n] = pk.u;
    }
    __builtin_amdgcn_wave_barrier();

    // O^T[p, c] += P~ (16x32, perm-K) x^T (32x16, perm-K) for 4 c-tiles
    const v16h Ap = load_A(stage[wave], 64, lane);
#pragma unroll
    for (int j = 0; j < 4; ++j)
      acc[j] = __builtin_amdgcn_wmma_f32_16x16x32_f16(false, Ap, false,
          load_B_ncontig(xt + j * 16 * 32, 32, 0, lane), (short)0,
          acc[j], false, false);
  }

  // normalize and restage O^T as f16 for the projection matmul
#pragma unroll
  for (int r = 0; r < 8; ++r) {
    const float inv = 1.0f / lrow[r];
    const int row = r + ((lane >> 4) << 3);
    const int n = lane & 15;
#pragma unroll
    for (int j = 0; j < 4; ++j) {
      acc[j][r] *= inv;
      stage[wave][row * 64 + j * 16 + n] = (_Float16)acc[j][r];
    }
  }
  __builtin_amdgcn_wave_barrier();

  const v16h Ao0 = load_A(stage[wave], 64, lane);
  const v16h Ao1 = load_A(stage[wave] + 32, 64, lane);
#pragma unroll
  for (int nt = 0; nt < 4; ++nt) {
    v8f d = (v8f){0.f,0.f,0.f,0.f,0.f,0.f,0.f,0.f};
    d = __builtin_amdgcn_wmma_f32_16x16x32_f16(false, Ao0, false,
        load_B_ncontig(ows + nt * 16 * Cn, Cn, 0, lane), (short)0, d, false, false);
    d = __builtin_amdgcn_wmma_f32_16x16x32_f16(false, Ao1, false,
        load_B_ncontig(ows + nt * 16 * Cn, Cn, 32, lane), (short)0, d, false, false);
#pragma unroll
    for (int r = 0; r < 8; ++r) {
      const int row = r + ((lane >> 4) << 3);
      const int o = nt * 16 + (lane & 15);
      const size_t gi = ((size_t)(b * Cn + o) * Tn + t) * Pn + (p0 + row);
      __builtin_nontemporal_store(d[r] + x_in[gi], &out[gi]);
    }
  }
}

// ---------------------------------------------------------------------------
extern "C" void kernel_launch(void* const* d_in, const int* in_sizes, int n_in,
                              void* d_out, int out_size, void* d_ws, size_t ws_size,
                              hipStream_t stream) {
  const float* x_in = (const float*)d_in[0];
  const float* tw   = (const float*)d_in[1];
  const float* pw   = (const float*)d_in[2];
  const float* ow   = (const float*)d_in[3];
  float* out = (float*)d_out;

  char* ws = (char*)d_ws;
  float*    g  = (float*)ws;                                   // 8*64*64 f32
  _Float16* xh = (_Float16*)(ws + (size_t)Tn * Cn * Cn * 4);   // 16*2304*64 f16
  _Float16* yh = xh + (size_t)SLn * Pn * Cn;

  g_kernel<<<Tn, 256, 0, stream>>>(tw, pw, g);
  xy_kernel<<<dim3(Pn / 256, SLn), 256, 0, stream>>>(x_in, g, xh, yh);
  flash_kernel<<<dim3(Pn / 128, SLn), 256, 0, stream>>>(xh, yh, x_in, ow, out);
}
